// DMoNLayer_5892695130194
// MI455X (gfx1250) — matrix operand order repaired and checked by reading, not compiled, not run on hardware
//
#include <hip/hip_runtime.h>
#include <hip/hip_bf16.h>

#define D_IN 128
#define D_H  256
#define KC   16

typedef __attribute__((ext_vector_type(8)))  float    v8f;
typedef __attribute__((ext_vector_type(4)))  _Float16 v4h;
typedef __attribute__((ext_vector_type(8)))  _Float16 v8h;
typedef __attribute__((ext_vector_type(16))) _Float16 v16h;

union V16 { v16h v; v8h h[2]; };

// X pitch: 136 halves = 272B = 68 dwords; 68 % 64 = 4 -> 16 rows land on bank
// offsets 4L; each 16B/lane A read sweeps all 64 banks exactly twice (minimum).
#define XP (D_IN + 8)
// h pitch: 264 halves = 528B = 132 dwords; 132 % 64 = 4 -> same property.
#define HP (D_H + 8)

// ---------------------------------------------------------------------------
// Weight prep: W1 -> W1t [256][128] f16 (transposed), W2 -> W2t [16][256] f16.
// B operands become 32 contiguous halves -> two global_load_b128 per lane.
// ---------------------------------------------------------------------------
__global__ __launch_bounds__(256) void dmon_prep_kernel(
    const float* __restrict__ W1, const float* __restrict__ W2,
    _Float16* __restrict__ W1t, _Float16* __restrict__ W2t)
{
    const int i = blockIdx.x * 256 + threadIdx.x;
    if (i < D_IN * D_H) {
        const int k = i >> 8;          // W1 row (K index), coalesced read on c
        const int c = i & (D_H - 1);
        W1t[(size_t)c * D_IN + k] = (_Float16)W1[i];
    }
    if (i < D_H * KC) {
        const int k = i >> 4;
        const int c = i & (KC - 1);
        W2t[(size_t)c * D_H + k] = (_Float16)W2[i];
    }
}

// ---------------------------------------------------------------------------
// Fused 2-layer MLP + row softmax, V_WMMA_F32_16X16X32_F16, fp32 accumulate.
// One block == 64 rows.  Layer 1: per wave 2 col-tiles x 4 row-tiles x 4
// k-steps = 32 WMMAs, A/B double-buffered across k-steps.  Layer 2: K=256
// split across 8 waves -> 1 WMMA per row-tile; fixed-order LDS reduction,
// shuffle softmax, coalesced store (plus optional f16 shadow of C).
// ---------------------------------------------------------------------------
__global__ __launch_bounds__(256) void dmon_mlp_kernel(
    const float* __restrict__ x,
    const _Float16* __restrict__ W1t, const float* __restrict__ b1,
    const _Float16* __restrict__ W2t, const float* __restrict__ b2,
    float* __restrict__ C, _Float16* __restrict__ Ch, int N)
{
    __shared__ __align__(16) union {
        _Float16 X[64][XP];      // 17.4 KB f16 x tile
        float    P[8][64][KC];   // 32 KB layer-2 partials (X dead by then)
    } sU;
    __shared__ __align__(16) _Float16 sH[64][HP];  // 33.8 KB f16 h tile

    const int rowbase = blockIdx.x * 64;
    const int t = threadIdx.x;
    const int lane = t & 31;
    const int wave = t >> 5;
    const int half = lane >> 4;   // lane-half: K 0-7 vs 8-15 (and +16)
    const int l16  = lane & 15;

    // ---- stage 64x128 x-tile into LDS as f16 (coalesced float4 reads) ----
    {
        const float4* gx = (const float4*)(x);
        #pragma unroll
        for (int j = 0; j < 8; ++j) {
            const int f = t + 256 * j;          // float4 index, lanes contiguous
            const int row = f >> 5;             // 32 float4 per row
            const int c4  = f & 31;
            int srcrow = rowbase + row;
            if (srcrow >= N) srcrow = N - 1;    // clamp; final store is guarded
            float4 u = gx[(size_t)srcrow * (D_IN / 4) + c4];
            v4h p;
            p[0] = (_Float16)u.x; p[1] = (_Float16)u.y;
            p[2] = (_Float16)u.z; p[3] = (_Float16)u.w;
            *(v4h*)&sU.X[row][c4 * 4] = p;      // one 8B DS store
        }
    }
    __syncthreads();

    // ---- phase A: h = relu(x @ W1 + b1) ----
    const int colbase = wave * 32;
    v8f acc[2][4];
    {
        const float bias0 = b1[colbase + l16];
        const float bias1 = b1[colbase + 16 + l16];
        #pragma unroll
        for (int rt = 0; rt < 4; ++rt)
            #pragma unroll
            for (int i = 0; i < 8; ++i) {
                acc[0][rt][i] = bias0;
                acc[1][rt][i] = bias1;
            }
    }
    {
        const _Float16* Bp0 = W1t + (size_t)(colbase + l16) * D_IN + half * 8;
        const _Float16* Bp1 = Bp0 + (size_t)16 * D_IN;

        // prologue: k-step 0 operands
        V16 b0c, b1c, ac[4];
        b0c.h[0] = *(const v8h*)(Bp0);
        b0c.h[1] = *(const v8h*)(Bp0 + 16);
        b1c.h[0] = *(const v8h*)(Bp1);
        b1c.h[1] = *(const v8h*)(Bp1 + 16);
        #pragma unroll
        for (int rt = 0; rt < 4; ++rt) {
            ac[rt].h[0] = *(const v8h*)&sU.X[rt * 16 + l16][half * 8];
            ac[rt].h[1] = *(const v8h*)&sU.X[rt * 16 + l16][16 + half * 8];
        }

        #pragma unroll
        for (int ks = 0; ks < 4; ++ks) {        // K = 32 per step
            V16 b0n, b1n, an[4];
            if (ks < 3) {                        // prefetch next k-step
                const int kb = (ks + 1) * 32;
                b0n.h[0] = *(const v8h*)(Bp0 + kb);
                b0n.h[1] = *(const v8h*)(Bp0 + kb + 16);
                b1n.h[0] = *(const v8h*)(Bp1 + kb);
                b1n.h[1] = *(const v8h*)(Bp1 + kb + 16);
                #pragma unroll
                for (int rt = 0; rt < 4; ++rt) {
                    an[rt].h[0] = *(const v8h*)&sU.X[rt * 16 + l16][kb + half * 8];
                    an[rt].h[1] = *(const v8h*)&sU.X[rt * 16 + l16][kb + 16 + half * 8];
                }
            }
            #pragma unroll
            for (int rt = 0; rt < 4; ++rt) {
                acc[0][rt] = __builtin_amdgcn_wmma_f32_16x16x32_f16(
                    false, ac[rt].v, false, b0c.v, (short)0, acc[0][rt], false, false);
                acc[1][rt] = __builtin_amdgcn_wmma_f32_16x16x32_f16(
                    false, ac[rt].v, false, b1c.v, (short)0, acc[1][rt], false, false);
            }
            if (ks < 3) {
                b0c = b0n; b1c = b1n;
                #pragma unroll
                for (int rt = 0; rt < 4; ++rt) ac[rt] = an[rt];
            }
        }
    }
    // ReLU + f16 convert, scatter C layout (VGPR r -> M=r / M=r+8) into sH
    #pragma unroll
    for (int tile = 0; tile < 2; ++tile)
        #pragma unroll
        for (int rt = 0; rt < 4; ++rt)
            #pragma unroll
            for (int r = 0; r < 8; ++r) {
                float v = acc[tile][rt][r];
                sH[rt * 16 + r + half * 8][colbase + tile * 16 + l16] =
                    (_Float16)(v > 0.f ? v : 0.f);
            }
    __syncthreads();   // retires all X reads -> union reusable as P

    // ---- phase B: logits = h @ W2 + b2; one K=32 WMMA per row-tile ----
    {
        const int kbase = wave * 32;
        const _Float16* Bw = W2t + (size_t)l16 * D_H + kbase + half * 8;
        V16 bw;
        bw.h[0] = *(const v8h*)(Bw);
        bw.h[1] = *(const v8h*)(Bw + 16);
        #pragma unroll
        for (int rt = 0; rt < 4; ++rt) {
            V16 ah;
            ah.h[0] = *(const v8h*)&sH[rt * 16 + l16][kbase + half * 8];
            ah.h[1] = *(const v8h*)&sH[rt * 16 + l16][kbase + 16 + half * 8];
            v8f pacc;
            #pragma unroll
            for (int i = 0; i < 8; ++i) pacc[i] = 0.f;
            pacc = __builtin_amdgcn_wmma_f32_16x16x32_f16(
                false, ah.v, false, bw.v, (short)0, pacc, false, false);
            #pragma unroll
            for (int r = 0; r < 8; ++r)
                sU.P[wave][rt * 16 + r + half * 8][l16] = pacc[r];
        }
    }
    __syncthreads();

    // ---- fixed-order partial reduction + softmax + store ----
    const int col = t & 15;
    #pragma unroll
    for (int rr = 0; rr < 4; ++rr) {
        const int row = rr * 16 + (t >> 4);   // 16-lane group == one row
        float logit = b2[col];
        #pragma unroll
        for (int w = 0; w < 8; ++w)
            logit += sU.P[w][row][col];

        float m = logit;
        #pragma unroll
        for (int mask = 1; mask < 16; mask <<= 1)
            m = fmaxf(m, __shfl_xor(m, mask));
        float e = __expf(logit - m);
        float s = e;
        #pragma unroll
        for (int mask = 1; mask < 16; mask <<= 1)
            s += __shfl_xor(s, mask);

        const int orow = rowbase + row;
        if (orow < N) {
            float c = e / s;
            C[(size_t)orow * KC + col] = c;
            if (Ch) Ch[(size_t)orow * KC + col] = (_Float16)c;  // shadow for edge gathers
        }
    }
}

// ---------------------------------------------------------------------------
// Edge modularity term: deterministic per-block partials (no float atomics).
// f16 variant gathers the 32B/row shadow (halves L2 gather bytes+requests);
// f32 variant is the fallback when d_ws is too small for the shadow.
// ---------------------------------------------------------------------------
#define EDGE_BLOCKS 2048

__device__ __forceinline__ void edge_block_reduce(float sum, float* partials) {
    #pragma unroll
    for (int mask = 16; mask >= 1; mask >>= 1)
        sum += __shfl_xor(sum, mask);
    __shared__ float wsum[8];
    if ((threadIdx.x & 31) == 0) wsum[threadIdx.x >> 5] = sum;
    __syncthreads();
    if (threadIdx.x == 0) {
        float b = 0.f;
        #pragma unroll
        for (int w = 0; w < 8; ++w) b += wsum[w];
        partials[blockIdx.x] = b;
    }
}

__global__ __launch_bounds__(256) void dmon_edge_f16_kernel(
    const int* __restrict__ ei, const _Float16* __restrict__ Ch,
    float* __restrict__ partials, int E)
{
    float sum = 0.f;
    const int stride = gridDim.x * blockDim.x;
    for (int e = blockIdx.x * blockDim.x + threadIdx.x; e < E; e += stride) {
        const int u = ei[e];
        const int v = ei[E + e];
        const v8h* cu = (const v8h*)(Ch + (size_t)u * KC);
        const v8h* cv = (const v8h*)(Ch + (size_t)v * KC);
        #pragma unroll
        for (int j = 0; j < 2; ++j) {
            v8h a = cu[j], b = cv[j];
            #pragma unroll
            for (int i = 0; i < 8; ++i)
                sum = fmaf((float)a[i], (float)b[i], sum);
        }
    }
    edge_block_reduce(sum, partials);
}

__global__ __launch_bounds__(256) void dmon_edge_f32_kernel(
    const int* __restrict__ ei, const float* __restrict__ C,
    float* __restrict__ partials, int E)
{
    float sum = 0.f;
    const int stride = gridDim.x * blockDim.x;
    for (int e = blockIdx.x * blockDim.x + threadIdx.x; e < E; e += stride) {
        const int u = ei[e];
        const int v = ei[E + e];
        const float4* cu = (const float4*)(C + (size_t)u * KC);
        const float4* cv = (const float4*)(C + (size_t)v * KC);
        #pragma unroll
        for (int j = 0; j < 4; ++j) {
            float4 a = cu[j], b = cv[j];
            sum = fmaf(a.x, b.x, sum);
            sum = fmaf(a.y, b.y, sum);
            sum = fmaf(a.z, b.z, sum);
            sum = fmaf(a.w, b.w, sum);
        }
    }
    edge_block_reduce(sum, partials);
}

__global__ __launch_bounds__(256) void dmon_reduce_kernel(
    const float* __restrict__ partials, float* __restrict__ loss, int E)
{
    float s = 0.f;
    for (int i = threadIdx.x; i < EDGE_BLOCKS; i += 256)
        s += partials[i];
    #pragma unroll
    for (int mask = 16; mask >= 1; mask >>= 1)
        s += __shfl_xor(s, mask);
    __shared__ float wsum[8];
    if ((threadIdx.x & 31) == 0) wsum[threadIdx.x >> 5] = s;
    __syncthreads();
    if (threadIdx.x == 0) {
        float tot = 0.f;
        #pragma unroll
        for (int w = 0; w < 8; ++w) tot += wsum[w];
        loss[0] = -tot / (float)E;
    }
}

extern "C" void kernel_launch(void* const* d_in, const int* in_sizes, int n_in,
                              void* d_out, int out_size, void* d_ws, size_t ws_size,
                              hipStream_t stream) {
    const float* x  = (const float*)d_in[0];
    const int*   ei = (const int*)d_in[1];     // edge_index [2, E]
    const float* W1 = (const float*)d_in[2];
    const float* b1 = (const float*)d_in[3];
    const float* W2 = (const float*)d_in[4];
    const float* b2 = (const float*)d_in[5];

    const int N = in_sizes[0] / D_IN;
    const int E = in_sizes[1] / 2;

    float* C    = (float*)d_out;               // [N, 16]
    float* loss = C + (size_t)N * KC;          // scalar, concatenated after C

    // d_ws layout: W1t (64 KB) | W2t (8 KB) | [Ch f16 shadow (N*32 B)] | partials
    char* ws = (char*)d_ws;
    _Float16* W1t = (_Float16*)ws;
    size_t off = (size_t)D_IN * D_H * 2;
    _Float16* W2t = (_Float16*)(ws + off);
    off += (size_t)D_H * KC * 2;
    const size_t ch_bytes = (size_t)N * KC * 2;
    const bool use_f16_edge = ws_size >= off + ch_bytes + EDGE_BLOCKS * 4;

    _Float16* Ch;
    float* partials;
    if (use_f16_edge) {
        Ch = (_Float16*)(ws + off);
        partials = (float*)(ws + off + ch_bytes);
    } else {
        Ch = nullptr;
        partials = (float*)(ws + off);
    }

    dmon_prep_kernel<<<(D_IN * D_H + 255) / 256, 256, 0, stream>>>(W1, W2, W1t, W2t);
    const int nblocks = (N + 63) / 64;
    dmon_mlp_kernel<<<nblocks, 256, 0, stream>>>(x, W1t, b1, W2t, b2, C, Ch, N);
    if (use_f16_edge)
        dmon_edge_f16_kernel<<<EDGE_BLOCKS, 256, 0, stream>>>(ei, Ch, partials, E);
    else
        dmon_edge_f32_kernel<<<EDGE_BLOCKS, 256, 0, stream>>>(ei, C, partials, E);
    dmon_reduce_kernel<<<1, 256, 0, stream>>>(partials, loss, E);
}
